// QuantModule_DiffAE_LoRA_79680233276214
// MI455X (gfx1250) — 8-bit matrix operand order repaired and checked by reading, compiled, not run on hardware
//
#include <hip/hip_runtime.h>
#include <hip/hip_bf16.h>

typedef int v8i __attribute__((ext_vector_type(8)));
typedef int v4i __attribute__((ext_vector_type(4)));
typedef int v2i __attribute__((ext_vector_type(2)));

#define T_DIM 16384
#define DIN   4096
#define DOUT  4096
#define RANK  32
#define QMAX  127.0f
#define EPSQ  1e-8f

__global__ __launch_bounds__(256)
void k_wprep(const float* __restrict__ W,
             const float* __restrict__ A,
             const float* __restrict__ B,
             signed char* __restrict__ qw,
             float* __restrict__ a_w)
{
    const int o  = blockIdx.x;
    const int t  = threadIdx.x;
    const int i0 = t * 16;

    __shared__ float Brow[RANK];
    __shared__ float red[256];

    if (t < RANK) Brow[t] = B[(size_t)o * RANK + t];
    __syncthreads();

    float acc[16];
    const float* wrow = W + (size_t)o * DIN + i0;
    #pragma unroll
    for (int c = 0; c < 16; ++c) acc[c] = wrow[c];

    for (int r = 0; r < RANK; ++r) {
        const float b = Brow[r];
        const float* arow = A + (size_t)r * DIN + i0;
        #pragma unroll
        for (int c = 0; c < 16; ++c) acc[c] = fmaf(b, arow[c], acc[c]);
    }

    float m = 0.0f;
    #pragma unroll
    for (int c = 0; c < 16; ++c) m = fmaxf(m, fabsf(acc[c]));
    red[t] = m;
    __syncthreads();
    for (int s = 128; s > 0; s >>= 1) {
        if (t < s) red[t] = fmaxf(red[t], red[t + s]);
        __syncthreads();
    }
    const float aw = red[0] + EPSQ;
    if (t == 0) a_w[o] = aw;

    const float inv = QMAX / aw;
    signed char q[16];
    #pragma unroll
    for (int c = 0; c < 16; ++c) {
        float v = fminf(fmaxf(acc[c] * inv, -QMAX), QMAX);
        q[c] = (signed char)__float2int_rn(v);
    }
    *(v4i*)(qw + (size_t)o * DIN + i0) = *(const v4i*)q;
}

__global__ __launch_bounds__(256)
void k_xq(const float* __restrict__ x,
          const float* __restrict__ a_x_p,
          signed char* __restrict__ qx)
{
    const float ax  = fmaxf(a_x_p[0], EPSQ);
    const float inv = QMAX / ax;
    const size_t i0 = ((size_t)blockIdx.x * 256 + threadIdx.x) * 16;
    const float* xp = x + i0;
    signed char q[16];
    #pragma unroll
    for (int c = 0; c < 16; ++c) {
        float v = fminf(fmaxf(xp[c] * inv, -QMAX), QMAX);
        q[c] = (signed char)__float2int_rn(v);
    }
    *(v4i*)(qx + i0) = *(const v4i*)q;
}

#define KSTAGE  128
#define ABYTES  (128 * KSTAGE)
#define BUFSZ   (2 * ABYTES)
__global__ __launch_bounds__(256)
void k_gemm(const signed char* __restrict__ qx,
            const signed char* __restrict__ qw,
            const float* __restrict__ a_w,
            const float* __restrict__ bias,
            const float* __restrict__ a_x_p,
            float* __restrict__ out)
{
    __shared__ __align__(16) signed char smem[2 * BUFSZ];

    const int tid  = threadIdx.x;
    const int lane = tid & 31;
    const int wave = tid >> 5;
    const int wm   = wave >> 1;
    const int wn   = wave & 1;
    const int half = lane >> 4;
    const int l16  = lane & 15;

    const int m0 = blockIdx.y * 128;
    const int n0 = blockIdx.x * 128;

    const v8i vzero = {0, 0, 0, 0, 0, 0, 0, 0};
    v8i acc[2][4];
    #pragma unroll
    for (int mt = 0; mt < 2; ++mt)
        #pragma unroll
        for (int nt = 0; nt < 4; ++nt) acc[mt][nt] = vzero;

    auto issue_stage = [&](int k0, int buf) {
        #pragma unroll
        for (int c = 0; c < 4; ++c) {
            const int chunk = tid + c * 256;
            const int row   = chunk >> 3;
            const int off   = (chunk & 7) << 4;
            const unsigned la = (unsigned)(uintptr_t)
                (smem + buf * BUFSZ + row * KSTAGE + off);
            const signed char* ga = qx + (size_t)(m0 + row) * DIN + k0 + off;
            asm volatile("global_load_async_to_lds_b128 %0, %1, off"
                         :: "v"(la), "v"(ga) : "memory");
            const unsigned lb = la + ABYTES;
            const signed char* gb = qw + (size_t)(n0 + row) * DIN + k0 + off;
            asm volatile("global_load_async_to_lds_b128 %0, %1, off"
                         :: "v"(lb), "v"(gb) : "memory");
        }
    };

    issue_stage(0, 0);

    const int NSTAGE = DIN / KSTAGE;
    for (int it = 0; it < NSTAGE; ++it) {
        const int buf = it & 1;
        const bool more = (it + 1) < NSTAGE;
        if (more) issue_stage((it + 1) * KSTAGE, buf ^ 1);

        if (more) asm volatile("s_wait_asynccnt 0x8" ::: "memory");
        else      asm volatile("s_wait_asynccnt 0x0" ::: "memory");
        __syncthreads();

        const signed char* aBase = smem + buf * BUFSZ;
        const signed char* bBase = aBase + ABYTES;

        #pragma unroll
        for (int ks = 0; ks < 2; ++ks) {
            v8i afrag[2];
            #pragma unroll
            for (int mt = 0; mt < 2; ++mt) {
                const signed char* ap = aBase +
                    (wm * 32 + mt * 16 + l16) * KSTAGE + ks * 64 + half * 16;
                v2i* av = (v2i*)&afrag[mt];
                #pragma unroll
                for (int c = 0; c < 4; ++c)
                    av[c] = *(const v2i*)(ap + (c & 1) * 8 + (c >> 1) * 32);
            }
            v8i bfrag[4];
            #pragma unroll
            for (int nt = 0; nt < 4; ++nt) {
                const signed char* bp = bBase +
                    (wn * 64 + nt * 16 + l16) * KSTAGE + ks * 64 + half * 16;
                v4i* bv = (v4i*)&bfrag[nt];
                bv[0] = *(const v4i*)(bp);
                bv[1] = *(const v4i*)(bp + 32);
            }
            #pragma unroll
            for (int mt = 0; mt < 2; ++mt)
                #pragma unroll
                for (int nt = 0; nt < 4; ++nt)
                    acc[mt][nt] = __builtin_amdgcn_wmma_i32_16x16x64_iu8(
                        true, afrag[mt], true, bfrag[nt], acc[mt][nt],
                        false, false);
        }
        __syncthreads();
    }

    const float ax = fmaxf(a_x_p[0], EPSQ);
    #pragma unroll
    for (int nt = 0; nt < 4; ++nt) {
        const int n   = n0 + wn * 64 + nt * 16 + l16;
        const float s = ax * a_w[n] * (1.0f / (QMAX * QMAX));
        const float b = bias[n];
        #pragma unroll
        for (int mt = 0; mt < 2; ++mt) {
            const int mb = m0 + wm * 32 + mt * 16 + half * 8;
            const int* ai = (const int*)&acc[mt][nt];
            #pragma unroll
            for (int v = 0; v < 8; ++v)
                out[(size_t)(mb + v) * DOUT + n] = (float)ai[v] * s + b;
        }
    }
}

extern "C" void kernel_launch(void* const* d_in, const int* in_sizes, int n_in,
                              void* d_out, int out_size, void* d_ws, size_t ws_size,
                              hipStream_t stream) {
    const float* x    = (const float*)d_in[0];
    const float* W    = (const float*)d_in[1];
    const float* lA   = (const float*)d_in[2];
    const float* lB   = (const float*)d_in[3];
    const float* bias = (const float*)d_in[4];
    const float* a_x  = (const float*)d_in[5];
    float* out = (float*)d_out;

    signed char* qx = (signed char*)d_ws;
    signed char* qw = qx + (size_t)T_DIM * DIN;
    float* a_w = (float*)(qw + (size_t)DOUT * DIN);

    k_wprep<<<DOUT, 256, 0, stream>>>(W, lA, lB, qw, a_w);

    const int xq_blocks = (int)(((size_t)T_DIM * DIN) / (256 * 16));
    k_xq<<<xq_blocks, 256, 0, stream>>>(x, a_x, qx);

    dim3 grid(DOUT / 128, T_DIM / 128);
    k_gemm<<<grid, 256, 0, stream>>>(qx, qw, a_w, bias, a_x, out);
}
